// MultiHeadAttentionQuantum_65481071400109
// MI455X (gfx1250) — matrix operand order repaired and checked by reading, not compiled, not run on hardware
//
#include <hip/hip_runtime.h>
#include <hip/hip_bf16.h>
#include <cmath>

// ---------------------------------------------------------------------------
// MultiHeadAttentionQuantum for MI455X (gfx1250, wave32, WMMA)
//
//  1) qkv_quantum : q/k/v = x @ W^T, then analytic quantum layer
//       c_i = cos(y_i + theta_i);  z = (c1c2c3, c0c1, c0c1c2, c0c1c2c3)
//  2) attn : flash attention, 16 rows per wave, all-register inner loop:
//       S^T tiles via V_WMMA_F32_16X16X4_F32 (K=4 == dk exact shape; operands
//       swapped so the C-layout output IS the f16 A-layout for PV),
//       p = v_exp_f32(score')  (scores bounded in ~[-2.9,2.9] -> raw hw exp2,
//       no max subtraction, no denorm guard needed),
//       PV via V_WMMA_F32_16X16X32_F16 with a ones-column in V so the
//       persistent f32 accumulator's column 4 carries the softmax row sums.
//  3) outproj : out = attn_out @ wo^T.
// ---------------------------------------------------------------------------

typedef __attribute__((ext_vector_type(2)))  float    v2f;
typedef __attribute__((ext_vector_type(8)))  float    v8f;
typedef __attribute__((ext_vector_type(8)))  _Float16 v8h;
typedef __attribute__((ext_vector_type(16))) _Float16 v16h;

#define BATCH 4
#define SEQ   2048
#define EMB   64
#define DK    4
#define NH    16
#define BHD   (BATCH * NH)  // 64 (batch*heads)
#define LOG2E 1.4426950408889634f

// raw v_exp_f32: safe because our inputs are bounded far inside (-126, 128)
static __device__ __forceinline__ float fast_exp2(float x)
{
#if __has_builtin(__builtin_amdgcn_exp2f)
    return __builtin_amdgcn_exp2f(x);
#else
    return exp2f(x);
#endif
}

// ---------------------------------------------------------------- kernel 1 --
__global__ __launch_bounds__(64)
void qkv_quantum_kernel(const float* __restrict__ x,
                        const float* __restrict__ wq,
                        const float* __restrict__ wk,
                        const float* __restrict__ wv,
                        const float* __restrict__ theta,
                        float* __restrict__ Qh, float* __restrict__ Kh,
                        _Float16* __restrict__ Vt)
{
    __shared__ float xrow[EMB];
    __shared__ float yrow[EMB];
    const int tok = blockIdx.x;       // 0..8191  (b = tok/SEQ, s = tok%SEQ)
    const int mat = blockIdx.y;       // 0=q 1=k 2=v
    const int e   = threadIdx.x;      // 0..63

    const float* w = (mat == 0) ? wq : (mat == 1) ? wk : wv;

    xrow[e] = x[(size_t)tok * EMB + e];
    __syncthreads();

    // y_e = dot(x_row, W[e,:])   (nn.Linear: y = x @ W.T)
    const float4* wr4 = (const float4*)(w + (size_t)e * EMB);
    const float4* xr4 = (const float4*)xrow;
    float acc = 0.f;
#pragma unroll
    for (int j = 0; j < EMB / 4; ++j) {
        const float4 a = xr4[j], b = wr4[j];
        acc = fmaf(a.x, b.x, acc);
        acc = fmaf(a.y, b.y, acc);
        acc = fmaf(a.z, b.z, acc);
        acc = fmaf(a.w, b.w, acc);
    }
    yrow[e] = acc;
    __syncthreads();

    // analytic quantum layer on this head's 4 angles (args ~N(0,1): fast cos)
    const int h = e >> 2, d = e & 3;
    float c[4];
#pragma unroll
    for (int i = 0; i < 4; ++i) c[i] = __cosf(yrow[h * 4 + i] + theta[i]);
    const float c01 = c[0] * c[1];
    float z;
    if      (d == 0) z = c[1] * c[2] * c[3];
    else if (d == 1) z = c01;
    else if (d == 2) z = c01 * c[2];
    else             z = c01 * c[2] * c[3];

    const int b  = tok / SEQ, s = tok % SEQ;
    const int bh = b * NH + h;
    if (mat == 0)      Qh[((size_t)bh * SEQ + s) * DK + d] = z;
    else if (mat == 1) Kh[((size_t)bh * SEQ + s) * DK + d] = z;
    else               Vt[((size_t)bh * DK + d) * SEQ + s] = (_Float16)z;
}

// ---------------------------------------------------------------- kernel 2 --
__global__ __launch_bounds__(256)
void attn_kernel(const float* __restrict__ Qh, const float* __restrict__ Kh,
                 const _Float16* __restrict__ Vt, float* __restrict__ O)
{
    const int bh   = blockIdx.y;          // 0..63
    const int wv   = threadIdx.x >> 5;    // wave in WG (8 waves)
    const int lane = threadIdx.x & 31;
    const int n    = lane & 15;
    const int hi   = lane >> 4;
    const int s0   = (blockIdx.x * 8 + wv) * 16;  // this wave's 16 rows

    const float*    Qb = Qh + (size_t)bh * SEQ * DK;
    const float*    Kb = Kh + (size_t)bh * SEQ * DK;
    const _Float16* Vb = Vt + (size_t)bh * DK * SEQ;

    // B operand of the transposed-score WMMA: Q^T (col s = s0+n, k = 2hi+j).
    // Fold 1/sqrt(dk)=0.5 and log2(e) so p = exp2(raw WMMA output).
    v2f qb = *(const v2f*)(Qb + (size_t)(s0 + n) * DK + 2 * hi);
    const float fold = 0.5f * LOG2E;
    qb.x *= fold; qb.y *= fold;

    // persistent PV accumulator in C layout: vgpr r <-> s row s0 + r + 8*hi,
    // col n = d (0..3 = output, 4 = softmax row sum via ones-column of V)
    v8f acc;
#pragma unroll
    for (int r = 0; r < 8; ++r) acc[r] = 0.f;

    // constant part of the V B-operand (lane col n): n==4 -> ones, n>4 -> 0
    v16h bv;
#pragma unroll
    for (int i = 0; i < 16; ++i)
        bv[i] = (n == DK) ? (_Float16)1.f : (_Float16)0.f;

#pragma unroll 2
    for (int tb = 0; tb < SEQ; tb += 32) {
        // two transposed 16x16 score subtiles: S^T = K_tile x Q_tile^T.
        // C-layout output: lane holds col s = s0+n, rows t = off + r + 8*hi —
        // exactly the f16 A-layout runs (8hi..8hi+7, 16+8hi..+7) for PV.
        v8f st[2];
#pragma unroll
        for (int tt = 0; tt < 2; ++tt) {
            const v2f ka =
                *(const v2f*)(Kb + (size_t)(tb + tt * 16 + n) * DK + 2 * hi);
            v8f z;
#pragma unroll
            for (int i = 0; i < 8; ++i) z[i] = 0.f;
            st[tt] = __builtin_amdgcn_wmma_f32_16x16x4_f32(
                false, ka, false, qb, (short)0, z, false, false);
        }

        // p = exp2(score') directly into the PV A operand (registers only)
        v16h ap;
#pragma unroll
        for (int j = 0; j < 8; ++j) {
            ap[j]     = (_Float16)fast_exp2(st[0][j]);
            ap[8 + j] = (_Float16)fast_exp2(st[1][j]);
        }

        // V tile (B layout): lane col n = d, K = 8*hi + j (j<8), 16 + 8*hi + (j-8) (j>=8) over t
        if (n < DK) {
            const v8h* vp = (const v8h*)(Vb + (size_t)n * SEQ + tb + 8 * hi);
            const v8h v0 = vp[0], v1 = vp[2];
#pragma unroll
            for (int i = 0; i < 8; ++i) { bv[i] = v0[i]; bv[8 + i] = v1[i]; }
        }

        acc = __builtin_amdgcn_wmma_f32_16x16x32_f16(false, ap, false, bv,
                                                     (short)0, acc,
                                                     false, false);
    }

    // epilogue: col 4 of acc holds the softmax denominators
    float* ob = O + (size_t)bh * SEQ * DK;
#pragma unroll
    for (int r = 0; r < 8; ++r) {
        const float lr = __shfl(acc[r], DK, 16);   // broadcast within half
        if (n < DK) {
            const int srow = s0 + r + 8 * hi;
            ob[(size_t)srow * DK + n] = acc[r] / lr;
        }
    }
}

// ---------------------------------------------------------------- kernel 3 --
__global__ __launch_bounds__(64)
void outproj_kernel(const float* __restrict__ O, const float* __restrict__ wo,
                    float* __restrict__ out)
{
    __shared__ float arow[EMB];
    const int tok = blockIdx.x;
    const int e   = threadIdx.x;
    const int b = tok >> 11, s = tok & (SEQ - 1);
    const int h = e >> 2,  d = e & 3;
    arow[e] = O[(((size_t)(b * NH + h) * SEQ) + s) * DK + d];
    __syncthreads();
    const float4* wr4 = (const float4*)(wo + (size_t)e * EMB);
    const float4* ar4 = (const float4*)arow;
    float acc = 0.f;
#pragma unroll
    for (int j = 0; j < EMB / 4; ++j) {
        const float4 a = ar4[j], w = wr4[j];
        acc = fmaf(a.x, w.x, acc);
        acc = fmaf(a.y, w.y, acc);
        acc = fmaf(a.z, w.z, acc);
        acc = fmaf(a.w, w.w, acc);
    }
    out[(size_t)tok * EMB + e] = acc;
}

// ------------------------------------------------------------------ launch --
extern "C" void kernel_launch(void* const* d_in, const int* in_sizes, int n_in,
                              void* d_out, int out_size, void* d_ws,
                              size_t ws_size, hipStream_t stream)
{
    (void)in_sizes; (void)n_in; (void)out_size; (void)ws_size;
    const float* x     = (const float*)d_in[0];
    const float* wq    = (const float*)d_in[1];
    const float* wk    = (const float*)d_in[2];
    const float* wv    = (const float*)d_in[3];
    const float* wo    = (const float*)d_in[4];
    const float* theta = (const float*)d_in[5];

    // workspace: Qh(2MB f32) | Kh(2MB f32) | Vt(1MB f16) | O(2MB f32) = 7MB
    char* ws = (char*)d_ws;
    float*    Qh = (float*)(ws);
    float*    Kh = (float*)(ws + (size_t)2 * 1024 * 1024);
    _Float16* Vt = (_Float16*)(ws + (size_t)4 * 1024 * 1024);
    float*    Og = (float*)(ws + (size_t)5 * 1024 * 1024);

    qkv_quantum_kernel<<<dim3(BATCH * SEQ, 3), 64, 0, stream>>>(
        x, wq, wk, wv, theta, Qh, Kh, Vt);
    attn_kernel<<<dim3(SEQ / (16 * 8), BHD), 256, 0, stream>>>(Qh, Kh, Vt, Og);
    outproj_kernel<<<BATCH * SEQ, 64, 0, stream>>>(Og, wo, (float*)d_out);
}
